// Attention_82540681494971
// MI455X (gfx1250) — matrix operand order repaired and checked by reading, not compiled, not run on hardware
//
#include <hip/hip_runtime.h>
#include <hip/hip_bf16.h>

typedef __attribute__((ext_vector_type(16))) _Float16 v16h;
typedef __attribute__((ext_vector_type(8)))  _Float16 v8h;
typedef __attribute__((ext_vector_type(8)))  float    v8f;

#define NPOS 4096   // H*W
#define DHEAD 16
#define NBH 16      // B*HEAD

// ---------------------------------------------------------------------------
// Kernel 1: QKV 1x1 conv.  x[b][c][p] (f32) -> Qf/Kf [bh][pos][16] f16,
// Vt [bh][16][pos] f16 (layouts chosen to match CDNA5 WMMA fragment striping).
// ---------------------------------------------------------------------------
__global__ __launch_bounds__(256) void k_qkv(const float* __restrict__ x,
                                             const float* __restrict__ w,
                                             const float* __restrict__ bias,
                                             _Float16* __restrict__ Qf,
                                             _Float16* __restrict__ Kf,
                                             _Float16* __restrict__ Vt) {
  __shared__ float xs[64][65];
  const int b  = blockIdx.x >> 6;
  const int p0 = (blockIdx.x & 63) << 6;
  const int t  = threadIdx.x;
  for (int idx = t; idx < 64 * 64; idx += 256) {
    int c = idx >> 6, p = idx & 63;
    xs[c][p] = x[((size_t)b * 64 + c) * NPOS + p0 + p];
  }
  __syncthreads();
  const int px  = t & 63;
  const int pos = p0 + px;
  for (int o = (t >> 6); o < 192; o += 4) {
    float acc = bias[o];
    const float* wr = w + o * 64;
#pragma unroll
    for (int c = 0; c < 64; ++c) acc += wr[c] * xs[c][px];
    const int oc = o & 63;
    const int h = oc >> 4, d = oc & 15;
    const size_t bh = (size_t)b * 4 + h;
    const _Float16 hv = (_Float16)acc;
    if (o < 64)        Qf[(bh * NPOS + pos) * DHEAD + d] = hv;
    else if (o < 128)  Kf[(bh * NPOS + pos) * DHEAD + d] = hv;
    else               Vt[(bh * DHEAD + d) * NPOS + pos] = hv;
  }
}

// ---------------------------------------------------------------------------
// Kernel 2 (pass 1): stream S = Q^T K with WMMA.  Each lane keeps PRIVATE
// online-softmax stats (max,sum) for the tile column it owns -> zero
// cross-lane traffic in the hot loop.  One merge at the end (64 bpermutes
// per wave).  Store Z[n] = rowmax + log(rowsum).
// ---------------------------------------------------------------------------
__global__ __launch_bounds__(256) void k_stats(const _Float16* __restrict__ Qf,
                                               const _Float16* __restrict__ Kf,
                                               float* __restrict__ Z) {
  const int lane = threadIdx.x & 31;
  const int wid  = (blockIdx.x << 3) + (threadIdx.x >> 5);
  const int bh   = wid >> 8;
  const int n0   = (wid & 255) << 4;
  const int hi   = lane >> 4, lj = lane & 15;

  // A fragment: row M = lj holds Q[d, n0+lj]; lanes<16 carry K(=d)0..15,
  // lanes>=16 are the zero-padded K 16..31.
  v16h a = {};
  {
    if (hi == 0) a = *(const v16h*)(Qf + ((size_t)bh * NPOS + n0 + lj) * DHEAD);
  }
  float m8[8], s8[8];
#pragma unroll
  for (int r = 0; r < 8; ++r) { m8[r] = -3.0e38f; s8[r] = 0.0f; }

  const _Float16* kbase = Kf + (size_t)bh * NPOS * DHEAD;
#pragma unroll 2
  for (int m0 = 0; m0 < NPOS; m0 += 16) {
    // B fragment: lanes<16 hold K(=d)0..15 of column m0+lane; lanes>=16 are
    // the zero-padded K 16..31 rows.
    v16h bk = {};
    if (lane < 16) bk = *(const v16h*)(kbase + (size_t)(m0 + lane) * DHEAD);
    v8f c = {};
    c = __builtin_amdgcn_wmma_f32_16x16x32_f16(false, a, false, bk,
                                               (short)0, c, false, false);
#pragma unroll
    for (int r = 0; r < 8; ++r) {
      // S[n0 + r + 8*hi][m0 + lj]; per-lane private online update.
      const float v  = c[r];
      const float nm = fmaxf(m8[r], v);
      s8[r] = s8[r] * __expf(m8[r] - nm) + __expf(v - nm);
      m8[r] = nm;
    }
  }
  // Merge the 16 per-lane partial (max,sum) pairs within each half-wave.
#pragma unroll
  for (int r = 0; r < 8; ++r) {
    float m = m8[r], s = s8[r];
#pragma unroll
    for (int mask = 1; mask <= 8; mask <<= 1) {
      const float om = __shfl_xor(m, mask);
      const float os = __shfl_xor(s, mask);
      const float nm = fmaxf(m, om);
      s = s * __expf(m - nm) + os * __expf(om - nm);
      m = nm;
    }
    if (lj == 0)
      Z[(size_t)bh * NPOS + n0 + hi * 8 + r] = m + __logf(s);
  }
}

// ---------------------------------------------------------------------------
// Kernel 3 (pass 2): O[:, m0..m0+15] = sum_n V[:, n] * exp(S[n,m]-Z[n]).
// Recompute S tiles with WMMA, normalize, LDS-transpose to B fragments,
// accumulate O with WMMA (K = 32 fully utilized on the n contraction).
// ---------------------------------------------------------------------------
__global__ __launch_bounds__(256) void k_attn(const _Float16* __restrict__ Qf,
                                              const _Float16* __restrict__ Kf,
                                              const _Float16* __restrict__ Vt,
                                              const float* __restrict__ Z,
                                              float* __restrict__ AO) {
  __shared__ alignas(64) _Float16 plds[8][16][32];   // per-wave 16(m)x32(n)
  const int lane = threadIdx.x & 31;
  const int w    = threadIdx.x >> 5;
  const int wid  = (blockIdx.x << 3) + w;
  const int bh   = wid >> 8;
  const int m0   = (wid & 255) << 4;
  const int hi   = lane >> 4, lj = lane & 15;

  v16h bk = {};                                      // K columns m0..m0+15 (fixed)
  if (lane < 16)
    bk = *(const v16h*)(Kf + ((size_t)bh * NPOS + m0 + lane) * DHEAD);

  const float* zr = Z + (size_t)bh * NPOS;
  v8f o = {};

  for (int n0 = 0; n0 < NPOS; n0 += 32) {
    // Two 16x16 S tiles covering rows n0..n0+31.
    v16h a1 = {}, a2 = {};
    if (hi == 0) {
      a1 = *(const v16h*)(Qf + ((size_t)bh * NPOS + n0 + lj) * DHEAD);
      a2 = *(const v16h*)(Qf + ((size_t)bh * NPOS + n0 + 16 + lj) * DHEAD);
    }
    const v8f zc = {};
    v8f s1 = __builtin_amdgcn_wmma_f32_16x16x32_f16(false, a1, false, bk,
                                                    (short)0, zc, false, false);
    v8f s2 = __builtin_amdgcn_wmma_f32_16x16x32_f16(false, a2, false, bk,
                                                    (short)0, zc, false, false);
    // Normalize: P = exp(S - Z[n]); pack to f16 and transpose via LDS into
    // B-fragment order (row j = output column, 32 contiguous k values).
    v8h p1, p2;
#pragma unroll
    for (int r = 0; r < 8; ++r) {
      p1[r] = (_Float16)__expf(s1[r] - zr[n0 + hi * 8 + r]);
      p2[r] = (_Float16)__expf(s2[r] - zr[n0 + 16 + hi * 8 + r]);
    }
    *(v8h*)(&plds[w][lj][hi * 8])      = p1;   // k = i = r + 8*hi
    *(v8h*)(&plds[w][lj][16 + hi * 8]) = p2;   // k = 16 + r + 8*hi
    asm volatile("s_wait_dscnt 0" ::: "memory");   // cross-lane LDS RAW
    v16h bp = *(const v16h*)(&plds[w][lj][hi * 16]);
    asm volatile("" ::: "memory");                 // keep next stores below
    // V A-fragment: row M = d = lj, K = n chunk (two contiguous 16B loads).
    v16h av;
    {
      const _Float16* vb = Vt + ((size_t)bh * DHEAD + lj) * NPOS + n0 + hi * 16;
      v8h v0 = *(const v8h*)(vb);
      v8h v1 = *(const v8h*)(vb + 8);
#pragma unroll
      for (int i = 0; i < 8; ++i) { av[i] = v0[i]; av[i + 8] = v1[i]; }
    }
    o = __builtin_amdgcn_wmma_f32_16x16x32_f16(false, av, false, bp,
                                               (short)0, o, false, false);
  }
  const int b = bh >> 2, h = bh & 3;
#pragma unroll
  for (int r = 0; r < 8; ++r) {
    const int cch = h * 16 + hi * 8 + r;           // channel = head*16 + d
    AO[((size_t)b * 64 + cch) * NPOS + m0 + lj] = o[r];
  }
}

// ---------------------------------------------------------------------------
// Kernel 4: output projection (f32).
// ---------------------------------------------------------------------------
__global__ __launch_bounds__(256) void k_proj(const float* __restrict__ AO,
                                              const float* __restrict__ w,
                                              const float* __restrict__ bias,
                                              float* __restrict__ out) {
  __shared__ float xs[64][65];
  const int b  = blockIdx.x >> 6;
  const int p0 = (blockIdx.x & 63) << 6;
  const int t  = threadIdx.x;
  for (int idx = t; idx < 64 * 64; idx += 256) {
    int c = idx >> 6, p = idx & 63;
    xs[c][p] = AO[((size_t)b * 64 + c) * NPOS + p0 + p];
  }
  __syncthreads();
  const int px = t & 63;
  for (int o = (t >> 6); o < 64; o += 4) {
    float acc = bias[o];
    const float* wr = w + o * 64;
#pragma unroll
    for (int c = 0; c < 64; ++c) acc += wr[c] * xs[c][px];
    out[((size_t)b * 64 + o) * NPOS + p0 + px] = acc;
  }
}

extern "C" void kernel_launch(void* const* d_in, const int* in_sizes, int n_in,
                              void* d_out, int out_size, void* d_ws, size_t ws_size,
                              hipStream_t stream) {
  const float* x      = (const float*)d_in[0];
  const float* qkv_w  = (const float*)d_in[1];
  const float* qkv_b  = (const float*)d_in[2];
  const float* proj_w = (const float*)d_in[3];
  const float* proj_b = (const float*)d_in[4];
  float* out = (float*)d_out;

  char* ws = (char*)d_ws;                       // total ~10.25 MB
  _Float16* Qf = (_Float16*)(ws);               // 2 MB  [16][4096][16]
  _Float16* Kf = (_Float16*)(ws + (2u << 20));  // 2 MB
  _Float16* Vt = (_Float16*)(ws + (4u << 20));  // 2 MB  [16][16][4096]
  float*    Z  = (float*)(ws + (6u << 20));                     // 256 KB
  float*    AO = (float*)(ws + (6u << 20) + (256u << 10));      // 4 MB

  k_qkv  <<<256, 256, 0, stream>>>(x, qkv_w, qkv_b, Qf, Kf, Vt);
  k_stats<<<512, 256, 0, stream>>>(Qf, Kf, Z);
  k_attn <<<512, 256, 0, stream>>>(Qf, Kf, Vt, Z, AO);
  k_proj <<<256, 256, 0, stream>>>(AO, proj_w, proj_b, out);
}
